// CrossAttentionFusion_15985868275837
// MI455X (gfx1250) — compile-verified, matrix operand order checked by reading, not run on hardware
//
#include <hip/hip_runtime.h>
#include <hip/hip_bf16.h>
#include <math.h>

#define GNN 512
#define RNN 1024
#define NB  65536
#define OUT_LD 1536
#define ROWS 32         // batch rows per workgroup
#define THREADS 256     // 8 wave32s: 2 row-groups x 4 n-groups

typedef __bf16 bf16;
typedef __attribute__((ext_vector_type(16))) __bf16 v16bf;
typedef __attribute__((ext_vector_type(4)))  __bf16 v4bf;
typedef __attribute__((ext_vector_type(8)))  float  v8f;

// exact pointer element type expected by the async-LDS builtins (per hipcc diag)
typedef int v4i __attribute__((vector_size(16)));
typedef __attribute__((address_space(1))) v4i as1_v4i;
typedef __attribute__((address_space(3))) v4i as3_v4i;

#if defined(__HIP_DEVICE_COMPILE__) && __has_builtin(__builtin_amdgcn_global_load_async_to_lds_b128)
#define HAVE_ASYNC_COPY 1
#else
#define HAVE_ASYNC_COPY 0
#endif

union FragU { v16bf v; uint4 q[2]; };

// 16x32 bf16 WMMA operand fragment (CDNA5 16-bit layout: lanes 0-15 hold
// K=kb+[0..7] / kb+16+[0..7]; lanes 16-31 hold K=kb+8+[0..7] / kb+24+[0..7];
// lane%16 indexes M (A, LDS rows) or N (B, weight rows); both K-contiguous).
__device__ __forceinline__ v16bf load_frag(const bf16* base, int ldk, int kb, int lane) {
    const int r = lane & 15;
    const int h = (lane >> 4) & 1;
    const bf16* p = base + (size_t)r * ldk + kb + h * 8;
    FragU f;
    f.q[0] = *(const uint4*)(p);
    f.q[1] = *(const uint4*)(p + 16);
    return f.v;
}

__device__ __forceinline__ v8f wmma_bf16(v16bf a, v16bf b, v8f c) {
    return __builtin_amdgcn_wmma_f32_16x16x32_bf16(false, a, false, b, (short)0, c, false, false);
}

__device__ __forceinline__ v8f zero8() {
    v8f z = {0.f, 0.f, 0.f, 0.f, 0.f, 0.f, 0.f, 0.f};
    return z;
}

// Async global->LDS copy of an fp32 vector (16B per lane per op, ASYNCcnt).
__device__ __forceinline__ void async_copy_f32(const float* __restrict__ g, float* l,
                                               int nfloats, int tid, int nthreads) {
#if HAVE_ASYNC_COPY
    for (int i = tid * 4; i < nfloats; i += nthreads * 4) {
        __builtin_amdgcn_global_load_async_to_lds_b128(
            (as1_v4i*)(g + i), (as3_v4i*)(l + i), 0, 0);
    }
#else
    for (int i = tid * 4; i < nfloats; i += nthreads * 4)
        *(float4*)(l + i) = *(const float4*)(g + i);
#endif
}

__device__ __forceinline__ void wait_async0() {
#if HAVE_ASYNC_COPY
#if __has_builtin(__builtin_amdgcn_s_wait_asynccnt)
    __builtin_amdgcn_s_wait_asynccnt(0);
#else
    asm volatile("s_wait_asynccnt 0" ::: "memory");
#endif
#endif
}

// ---------------- fp32 -> bf16 cast (weights) ----------------
__global__ void caf_cast_bf16(const float* __restrict__ x, bf16* __restrict__ y, int n4) {
    int i = blockIdx.x * blockDim.x + threadIdx.x;
    if (i < n4) {
        float4 f = ((const float4*)x)[i];
        v4bf o = { (bf16)f.x, (bf16)f.y, (bf16)f.z, (bf16)f.w };
        ((v4bf*)y)[i] = o;
    }
}

// ---------------- weight fold: C(N,K) = A(N,J) @ B(J,K), bf16 out ----------------
__global__ void caf_fold_w(const float* __restrict__ A, const float* __restrict__ B,
                           bf16* __restrict__ C, int J, int K) {
    __shared__ float As[16][16];
    __shared__ float Bs[16][17];
    const int tx = threadIdx.x, ty = threadIdx.y;
    const int kBase = blockIdx.x * 16, nBase = blockIdx.y * 16;
    float acc = 0.f;
    for (int j0 = 0; j0 < J; j0 += 16) {
        As[ty][tx] = A[(size_t)(nBase + ty) * J + j0 + tx];
        Bs[ty][tx] = B[(size_t)(j0 + ty) * K + kBase + tx];
        __syncthreads();
#pragma unroll
        for (int jj = 0; jj < 16; ++jj) acc += As[ty][jj] * Bs[jj][tx];
        __syncthreads();
    }
    C[(size_t)(nBase + ty) * K + kBase + tx] = (bf16)acc;
}

// ---------------- bias fold: bc[n] = sum_j A[n,j]*b[j] ----------------
__global__ void caf_fold_b(const float* __restrict__ A, const float* __restrict__ b,
                           float* __restrict__ bc, int N, int J) {
    int n = blockIdx.x * blockDim.x + threadIdx.x;
    if (n < N) {
        float acc = 0.f;
        const float* row = A + (size_t)n * J;
        for (int j = 0; j < J; j += 4) {
            float4 a = *(const float4*)(row + j);
            float4 v = *(const float4*)(b + j);
            acc += a.x * v.x + a.y * v.y + a.z * v.z + a.w * v.w;
        }
        bc[n] = acc;
    }
}

// ---------------- fused cross-attention: 32 rows per workgroup ----------------
__global__ __launch_bounds__(THREADS)
void caf_fused(const float* __restrict__ h_gnn, const float* __restrict__ h_rnn,
               const float* __restrict__ bq_gnn, const float* __restrict__ bk_rnn,
               const float* __restrict__ bq_rnn, const float* __restrict__ bk_gnn,
               const bf16*  __restrict__ Wq_g,   const bf16*  __restrict__ Wk_r,
               const bf16*  __restrict__ Wq_r,   const bf16*  __restrict__ Wk_g,
               const bf16*  __restrict__ Wc_r2g, const bf16*  __restrict__ Wc_g2r,
               const float* __restrict__ bc_r2g, const float* __restrict__ bc_g2r,
               float* __restrict__ out) {
    __shared__ __align__(16) bf16 ldsXg[ROWS * GNN];   // 32 KB
    __shared__ __align__(16) bf16 ldsXr[ROWS * RNN];   // 64 KB
    __shared__ __align__(16) float lbq_g[GNN], lbk_r[GNN];        // biases staged async
    __shared__ __align__(16) float lbq_r[RNN], lbk_g[RNN];
    __shared__ __align__(16) float lbc_r2g[GNN], lbc_g2r[RNN];
    __shared__ float redA[128], redB[128];             // [rowgrp][ngrp][16 rows]
    __shared__ float wgv[ROWS], wrv[ROWS];

    const int tid  = threadIdx.x;
    const int lane = tid & 31;
    const int w    = tid >> 5;          // 8 wave32s
    const int ng   = w & 3;             // n-tile group
    const int rg   = w >> 2;            // row group (0/1)
    const size_t rowBase = (size_t)blockIdx.x * ROWS;

    // ---- kick off async bias staging (ASYNCcnt), overlap with conversion ----
    async_copy_f32(bq_gnn,  lbq_g,   GNN, tid, THREADS);
    async_copy_f32(bk_rnn,  lbk_r,   GNN, tid, THREADS);
    async_copy_f32(bq_rnn,  lbq_r,   RNN, tid, THREADS);
    async_copy_f32(bk_gnn,  lbk_g,   RNN, tid, THREADS);
    async_copy_f32(bc_r2g,  lbc_r2g, GNN, tid, THREADS);
    async_copy_f32(bc_g2r,  lbc_g2r, RNN, tid, THREADS);

    // ---- stage 32 activation rows to LDS as bf16 (HBM read exactly once) ----
    {
        const float4* sg = (const float4*)(h_gnn + rowBase * GNN);
        for (int i = tid; i < ROWS * GNN / 4; i += THREADS) {
            float4 f = sg[i];
            v4bf o = { (bf16)f.x, (bf16)f.y, (bf16)f.z, (bf16)f.w };
            *(v4bf*)(ldsXg + 4 * i) = o;
        }
        const float4* sr = (const float4*)(h_rnn + rowBase * RNN);
        for (int i = tid; i < ROWS * RNN / 4; i += THREADS) {
            float4 f = sr[i];
            v4bf o = { (bf16)f.x, (bf16)f.y, (bf16)f.z, (bf16)f.w };
            *(v4bf*)(ldsXr + 4 * i) = o;
        }
    }
    wait_async0();
    __syncthreads();

    const bf16* Ag = ldsXg + (size_t)rg * 16 * GNN;   // this wave's 16 rows
    const bf16* Ar = ldsXr + (size_t)rg * 16 * RNN;

    // ---- s_g[m] = (Xg Wq^T + bq) . (Xr Wk^T + bk) over 512 cols ----
    {
        v8f s = zero8();
        for (int nt = ng; nt < GNN / 16; nt += 4) {
            v8f qa = zero8(), ka = zero8();
            const bf16* wq = Wq_g + (size_t)nt * 16 * GNN;
            const bf16* wk = Wk_r + (size_t)nt * 16 * RNN;
            __builtin_prefetch(wq + 16 * GNN, 0, 0);
            for (int kb = 0; kb < GNN; kb += 32)
                qa = wmma_bf16(load_frag(Ag, GNN, kb, lane), load_frag(wq, GNN, kb, lane), qa);
            for (int kb = 0; kb < RNN; kb += 32)
                ka = wmma_bf16(load_frag(Ar, RNN, kb, lane), load_frag(wk, RNN, kb, lane), ka);
            const int col = nt * 16 + (lane & 15);
            const float bq = lbq_g[col], bk = lbk_r[col];
#pragma unroll
            for (int v = 0; v < 8; ++v) s[v] += (qa[v] + bq) * (ka[v] + bk);
        }
#pragma unroll
        for (int v = 0; v < 8; ++v) {   // reduce over the 16 N-lanes of each half
            float x = s[v];
            x += __shfl_xor(x, 1); x += __shfl_xor(x, 2);
            x += __shfl_xor(x, 4); x += __shfl_xor(x, 8);
            if ((lane & 15) == 0) redA[rg * 64 + ng * 16 + v + 8 * (lane >> 4)] = x;
        }
    }
    // ---- s_r[m] = (Xr Wq^T + bq) . (Xg Wk^T + bk) over 1024 cols ----
    {
        v8f s = zero8();
        for (int nt = ng; nt < RNN / 16; nt += 4) {
            v8f qa = zero8(), ka = zero8();
            const bf16* wq = Wq_r + (size_t)nt * 16 * RNN;
            const bf16* wk = Wk_g + (size_t)nt * 16 * GNN;
            __builtin_prefetch(wq + 16 * RNN, 0, 0);
            for (int kb = 0; kb < RNN; kb += 32)
                qa = wmma_bf16(load_frag(Ar, RNN, kb, lane), load_frag(wq, RNN, kb, lane), qa);
            for (int kb = 0; kb < GNN; kb += 32)
                ka = wmma_bf16(load_frag(Ag, GNN, kb, lane), load_frag(wk, GNN, kb, lane), ka);
            const int col = nt * 16 + (lane & 15);
            const float bq = lbq_r[col], bk = lbk_g[col];
#pragma unroll
            for (int v = 0; v < 8; ++v) s[v] += (qa[v] + bq) * (ka[v] + bk);
        }
#pragma unroll
        for (int v = 0; v < 8; ++v) {
            float x = s[v];
            x += __shfl_xor(x, 1); x += __shfl_xor(x, 2);
            x += __shfl_xor(x, 4); x += __shfl_xor(x, 8);
            if ((lane & 15) == 0) redB[rg * 64 + ng * 16 + v + 8 * (lane >> 4)] = x;
        }
    }
    __syncthreads();

    // ---- per-row sigmoid gates + 2-way softmax importance ----
    if (tid < ROWS) {
        const int g = (tid >> 4) * 64 + (tid & 15);
        float sg = redA[g] + redA[g + 16] + redA[g + 32] + redA[g + 48];
        float sr = redB[g] + redB[g + 16] + redB[g + 32] + redB[g + 48];
        float wg = 1.0f / (1.0f + __expf(-sg * 0.04419417382415922f)); // 1/sqrt(512)
        float wr = 1.0f / (1.0f + __expf(-sr * 0.03125f));             // 1/sqrt(1024)
        wgv[tid] = wg; wrv[tid] = wr;
        float eg = __expf(wg), er = __expf(wr);
        float inv = 1.0f / (eg + er);
        size_t row = rowBase + tid;
        out[(size_t)NB * OUT_LD + row * 2 + 0] = eg * inv;
        out[(size_t)NB * OUT_LD + row * 2 + 1] = er * inv;
    }
    __syncthreads();

    // ---- h_fused[:, 0:512] = h_gnn + wg * (Xr @ Wc_r2g^T + bc_r2g) ----
    for (int nt = ng; nt < GNN / 16; nt += 4) {
        v8f acc = zero8();
        const bf16* wc = Wc_r2g + (size_t)nt * 16 * RNN;
        __builtin_prefetch(wc + 16 * RNN, 0, 0);
        for (int kb = 0; kb < RNN; kb += 32)
            acc = wmma_bf16(load_frag(Ar, RNN, kb, lane), load_frag(wc, RNN, kb, lane), acc);
        const int col = nt * 16 + (lane & 15);
        const float bias = lbc_r2g[col];
#pragma unroll
        for (int v = 0; v < 8; ++v) {
            const int r = rg * 16 + v + 8 * (lane >> 4);
            const size_t row = rowBase + r;
            out[row * OUT_LD + col] = h_gnn[row * GNN + col] + wgv[r] * (acc[v] + bias);
        }
    }
    // ---- h_fused[:, 512:1536] = h_rnn + wr * (Xg @ Wc_g2r^T + bc_g2r) ----
    for (int nt = ng; nt < RNN / 16; nt += 4) {
        v8f acc = zero8();
        const bf16* wc = Wc_g2r + (size_t)nt * 16 * GNN;
        __builtin_prefetch(wc + 16 * GNN, 0, 0);
        for (int kb = 0; kb < GNN; kb += 32)
            acc = wmma_bf16(load_frag(Ag, GNN, kb, lane), load_frag(wc, GNN, kb, lane), acc);
        const int col = nt * 16 + (lane & 15);
        const float bias = lbc_g2r[col];
#pragma unroll
        for (int v = 0; v < 8; ++v) {
            const int r = rg * 16 + v + 8 * (lane >> 4);
            const size_t row = rowBase + r;
            out[row * OUT_LD + GNN + col] = h_rnn[row * RNN + col] + wrv[r] * (acc[v] + bias);
        }
    }
}

extern "C" void kernel_launch(void* const* d_in, const int* in_sizes, int n_in,
                              void* d_out, int out_size, void* d_ws, size_t ws_size,
                              hipStream_t stream) {
    const float* h_gnn  = (const float*)d_in[0];
    const float* h_rnn  = (const float*)d_in[1];
    const float* Wq_gnn = (const float*)d_in[2];
    const float* bq_gnn = (const float*)d_in[3];
    const float* Wk_rnn = (const float*)d_in[4];
    const float* bk_rnn = (const float*)d_in[5];
    const float* Wq_rnn = (const float*)d_in[6];
    const float* bq_rnn = (const float*)d_in[7];
    const float* Wk_gnn = (const float*)d_in[8];
    const float* bk_gnn = (const float*)d_in[9];
    const float* Wv_gnn = (const float*)d_in[10];
    const float* bv_gnn = (const float*)d_in[11];
    const float* Wv_rnn = (const float*)d_in[12];
    const float* bv_rnn = (const float*)d_in[13];
    const float* W_r2g  = (const float*)d_in[14];
    const float* W_g2r  = (const float*)d_in[15];
    float* out = (float*)d_out;

    // workspace carve (bf16 weight copies + folded weights/biases, ~6.5 MB)
    char* p = (char*)d_ws;
    bf16*  WqB   = (bf16*)p;  p += (size_t)GNN * GNN * 2;
    bf16*  WkB   = (bf16*)p;  p += (size_t)GNN * RNN * 2;
    bf16*  WqrB  = (bf16*)p;  p += (size_t)RNN * RNN * 2;
    bf16*  WkgB  = (bf16*)p;  p += (size_t)RNN * GNN * 2;
    bf16*  WcR2G = (bf16*)p;  p += (size_t)GNN * RNN * 2;
    bf16*  WcG2R = (bf16*)p;  p += (size_t)RNN * GNN * 2;
    float* bcR2G = (float*)p; p += (size_t)GNN * 4;
    float* bcG2R = (float*)p; p += (size_t)RNN * 4;

    // 1) cast q/k weights to bf16
    caf_cast_bf16<<<(GNN * GNN / 4 + 255) / 256, 256, 0, stream>>>(Wq_gnn, WqB,  GNN * GNN / 4);
    caf_cast_bf16<<<(GNN * RNN / 4 + 255) / 256, 256, 0, stream>>>(Wk_rnn, WkB,  GNN * RNN / 4);
    caf_cast_bf16<<<(RNN * RNN / 4 + 255) / 256, 256, 0, stream>>>(Wq_rnn, WqrB, RNN * RNN / 4);
    caf_cast_bf16<<<(RNN * GNN / 4 + 255) / 256, 256, 0, stream>>>(Wk_gnn, WkgB, RNN * GNN / 4);

    // 2) fold value+projection weights: Wc_r2g = W_r2g @ Wv_rnn, Wc_g2r = W_g2r @ Wv_gnn
    caf_fold_w<<<dim3(RNN / 16, GNN / 16), dim3(16, 16), 0, stream>>>(W_r2g, Wv_rnn, WcR2G, RNN, RNN);
    caf_fold_w<<<dim3(GNN / 16, RNN / 16), dim3(16, 16), 0, stream>>>(W_g2r, Wv_gnn, WcG2R, GNN, GNN);
    caf_fold_b<<<(GNN + 255) / 256, 256, 0, stream>>>(W_r2g, bv_rnn, bcR2G, GNN, RNN);
    caf_fold_b<<<(RNN + 255) / 256, 256, 0, stream>>>(W_g2r, bv_gnn, bcG2R, RNN, GNN);

    // 3) fused cross-attention over 32-row tiles
    caf_fused<<<NB / ROWS, THREADS, 0, stream>>>(h_gnn, h_rnn,
                                                 bq_gnn, bk_rnn, bq_rnn, bk_gnn,
                                                 WqB, WkB, WqrB, WkgB,
                                                 WcR2G, WcG2R, bcR2G, bcG2R, out);
    (void)in_sizes; (void)n_in; (void)out_size; (void)ws_size;
}
